// SpatialAttentionModel_12833362280582
// MI455X (gfx1250) — matrix operand order repaired and checked by reading, not compiled, not run on hardware
//
#include <hip/hip_runtime.h>

// ---------------------------------------------------------------------------
// SpatialAttention for MI455X (gfx1250, wave32, WMMA bf16 16x16x32)
// B=8 T=12 N=512 D=64, 8 heads x d=8.
// Attention computed TRANSPOSED (S^T = K Q^T, O^T = V^T P^T) so softmax
// reductions are in-lane; row-sum folded into WMMA via a ones-row in V^T.
// All fragment traffic is b128 loads/stores from pre-swizzled LDS layouts.
// ---------------------------------------------------------------------------

typedef __attribute__((ext_vector_type(16))) __bf16 v16bf;
typedef __attribute__((ext_vector_type(8)))  __bf16 v8bf;   // 16 bytes
typedef __attribute__((ext_vector_type(2)))  __bf16 v2bf;   // 4 bytes
typedef __attribute__((ext_vector_type(8)))  float  v8f;

#define NROWS  49152   // B*T*N
#define DMODEL 64
#define NHEAD  8
#define DHEAD  8
#define NSEQ   512
#define NBT    96      // B*T

// 16-bit A fragment (16x32, MxK), lane half h, element e:
//   e<8 : K = 8h+e ; e>=8: K = 16+8h+(e-8)   (two contiguous runs of 8)
// 16-bit B fragment (32x16, KxN), lane half h, element e: same K order as A (8h+e ; 16+8h+(e-8))
// C/D f32: VGPR r: row = r + 8h, col = lane&15.

__device__ __forceinline__ v8f wmma_bf16(v16bf a, v16bf b, v8f c) {
  return __builtin_amdgcn_wmma_f32_16x16x32_bf16(
      false, a, false, b, (short)0, c, false, false);
}

__device__ __forceinline__ v16bf join8(v8bf lo, v8bf hi) {
  return __builtin_shufflevector(lo, hi, 0, 1, 2, 3, 4, 5, 6, 7,
                                 8, 9, 10, 11, 12, 13, 14, 15);
}

// ---------------------------------------------------------------------------
// Kernel 1: out[head][bt][n][d] = relu([X|STE] @ W + b)  (bf16 output)
// ---------------------------------------------------------------------------
__global__ __launch_bounds__(256) void qkv_proj_kernel(
    const float* __restrict__ X, const float* __restrict__ STE,
    const float* __restrict__ W, const float* __restrict__ bias,
    __bf16* __restrict__ out) {
  __shared__ v16bf sWfrag[4][4][32];   // [kblock][ntile][lane] : 16KB
  const int tid = threadIdx.x;
  {
    __bf16* wp = (__bf16*)sWfrag;
    for (int idx = tid; idx < 4 * 4 * 32; idx += 256) {
      const int ln = idx & 31, nt = (idx >> 5) & 3, kb = idx >> 7;
      const int hd = ln >> 4, lr = ln & 15;
      const int col = nt * 16 + lr;
      #pragma unroll
      for (int e = 0; e < 16; ++e)   // B-frag: K = kb*32 + (e<8 ? 8*hd+e : 16+8*hd+(e-8))
        wp[idx * 16 + e] = (__bf16)W[(kb * 32 + (e < 8 ? 8 * hd + e : 8 + 8 * hd + e)) * DMODEL + col];
    }
  }
  __syncthreads();

  const int lane = tid & 31, wave = tid >> 5;
  const int h = lane >> 4, lrow = lane & 15;
  const int row0 = (blockIdx.x * 8 + wave) * 16;
  const int arow = row0 + lrow;

  v8f acc[4] = {{}, {}, {}, {}};
  #pragma unroll
  for (int kb = 0; kb < 4; ++kb) {
    const int k0 = kb * 32;
    const float* src = (k0 < 64) ? X : STE;   // H = [X | STE]
    const int cb = k0 & 63;
    const float* rowp = src + (size_t)arow * DMODEL + cb;
    const float4 x0 = *(const float4*)(rowp + 8 * h);
    const float4 x1 = *(const float4*)(rowp + 8 * h + 4);
    const float4 y0 = *(const float4*)(rowp + 16 + 8 * h);
    const float4 y1 = *(const float4*)(rowp + 20 + 8 * h);
    v16bf a;
    a[0] = (__bf16)x0.x; a[1] = (__bf16)x0.y; a[2] = (__bf16)x0.z; a[3] = (__bf16)x0.w;
    a[4] = (__bf16)x1.x; a[5] = (__bf16)x1.y; a[6] = (__bf16)x1.z; a[7] = (__bf16)x1.w;
    a[8] = (__bf16)y0.x; a[9] = (__bf16)y0.y; a[10] = (__bf16)y0.z; a[11] = (__bf16)y0.w;
    a[12] = (__bf16)y1.x; a[13] = (__bf16)y1.y; a[14] = (__bf16)y1.z; a[15] = (__bf16)y1.w;
    #pragma unroll
    for (int nt = 0; nt < 4; ++nt)
      acc[nt] = wmma_bf16(a, sWfrag[kb][nt][lane], acc[nt]);
  }
  #pragma unroll
  for (int nt = 0; nt < 4; ++nt) {
    const int col = nt * 16 + lrow;
    const int head = col >> 3, dd = col & 7;
    const float bb = bias[col];
    #pragma unroll
    for (int r = 0; r < 8; ++r) {
      const int row = row0 + r + 8 * h;
      const int bt = row >> 9, n = row & 511;
      const float v = fmaxf(acc[nt][r] + bb, 0.0f);
      out[((size_t)(head * NBT + bt) * NSEQ + n) * DHEAD + dd] = (__bf16)v;
    }
  }
}

// ---------------------------------------------------------------------------
// Kernel 2: transposed flash attention per (head, bt). Block = 256, grid = 768.
// sKp: [key][32] cols 8..31 zero  -> K A-fragment = two contiguous 16B reads
// sVt: [16][512] V^T; row 8 = ones (folds row-sum l into the WMMA), 9..15 zero
// sPt: per-wave [query][key] P^T staging: b128 stores AND b128 loads
// ---------------------------------------------------------------------------
__global__ __launch_bounds__(256) void attn_kernel(
    const __bf16* __restrict__ Q, const __bf16* __restrict__ Km,
    const __bf16* __restrict__ V, __bf16* __restrict__ AO) {
  __shared__ __bf16 sKp[NSEQ][32];       // 32KB
  __shared__ __bf16 sVt[16][NSEQ];       // 16KB
  __shared__ __bf16 sPt[8][16][32];      // 8KB

  const int tid = threadIdx.x;
  const int head = blockIdx.x / NBT, bt = blockIdx.x % NBT;
  const size_t base = (size_t)(head * NBT + bt) * NSEQ * DHEAD;

  // K: one b128 load + four b128 LDS stores per key (data + zero padding)
  {
    const v8bf zero8 = {};
    for (int key = tid; key < NSEQ; key += 256) {
      const v8bf kv = ((const v8bf*)(Km + base))[key];
      *(v8bf*)&sKp[key][0]  = kv;
      *(v8bf*)&sKp[key][8]  = zero8;
      *(v8bf*)&sKp[key][16] = zero8;
      *(v8bf*)&sKp[key][24] = zero8;
    }
  }
  // V^T: two b128 loads per key-pair, transpose as packed 2xbf16 b32 stores
  {
    v2bf one2, zero2;
    one2[0] = (__bf16)1.0f; one2[1] = (__bf16)1.0f;
    zero2[0] = (__bf16)0.0f; zero2[1] = (__bf16)0.0f;
    for (int kk = tid; kk < NSEQ / 2; kk += 256) {
      const int key = kk * 2;
      const v8bf v0 = ((const v8bf*)(V + base))[key];
      const v8bf v1 = ((const v8bf*)(V + base))[key + 1];
      #pragma unroll
      for (int d = 0; d < 8; ++d) {
        v2bf p; p[0] = v0[d]; p[1] = v1[d];
        *(v2bf*)&sVt[d][key] = p;
      }
      *(v2bf*)&sVt[8][key] = one2;     // ones-row: accumulates l in the WMMA
      #pragma unroll
      for (int d = 9; d < 16; ++d) *(v2bf*)&sVt[d][key] = zero2;
    }
  }
  __syncthreads();

  const int lane = tid & 31, wave = tid >> 5;
  const int h = lane >> 4, lrow = lane & 15;
  const float scale = 0.35355339059327373f;   // 1/sqrt(8)

  for (int t = wave; t < NSEQ / 16; t += 8) {   // 16 query COLUMNS per tile
    const int qc0 = t * 16;
    // Q^T B-fragment (constant over K loop): lane = query col; 1/sqrt(d) folded in
    const v8bf qv = ((const v8bf*)(Q + base))[qc0 + lrow];
    v16bf bq = {};
    #pragma unroll
    for (int e = 0; e < 8; ++e)
      bq[e] = (h == 0) ? (__bf16)((float)qv[e] * scale) : (__bf16)0.0f;

    float m = -1e30f;     // per-lane running max (one query per lane)
    v8f o = {};           // O^T tile: VGPR r -> d = r+8h ; d=8 accumulates l

    for (int kb = 0; kb < NSEQ; kb += 32) {
      // ---- S^T = K @ Q^T : keys in C rows, queries in C lanes ----
      const v8bf* kr0 = (const v8bf*)&sKp[kb + lrow][0];
      const v8bf* kr1 = (const v8bf*)&sKp[kb + 16 + lrow][0];
      const v16bf ak0 = join8(kr0[h], kr0[2 + h]);
      const v16bf ak1 = join8(kr1[h], kr1[2 + h]);
      v8f z0 = {}, z1 = {};
      const v8f st0 = wmma_bf16(ak0, bq, z0);   // keys kb+8h+r
      const v8f st1 = wmma_bf16(ak1, bq, z1);   // keys kb+16+8h+r

      // ---- in-lane softmax over 16 key-values + one half-swap shfl ----
      float tmax = fmaxf(st0[0], st1[0]);
      #pragma unroll
      for (int r = 1; r < 8; ++r) tmax = fmaxf(tmax, fmaxf(st0[r], st1[r]));
      tmax = fmaxf(tmax, __shfl_xor(tmax, 16, 32));   // merge the two key halves
      const float mn = fmaxf(m, tmax);
      const float corr = __expf(m - mn);
      m = mn;
      v8bf p0, p1;
      #pragma unroll
      for (int r = 0; r < 8; ++r) {
        p0[r] = (__bf16)__expf(st0[r] - mn);
        p1[r] = (__bf16)__expf(st1[r] - mn);
      }
      #pragma unroll
      for (int r = 0; r < 8; ++r) o[r] *= corr;   // also rescales l (col d=8)

      // P^T staging: per-lane contiguous key runs -> two ds_store_b128
      *(v8bf*)&sPt[wave][lrow][8 * h]      = p0;   // keys kb+8h+0..7
      *(v8bf*)&sPt[wave][lrow][16 + 8 * h] = p1;   // keys kb+16+8h+0..7

      // ---- O^T += V^T @ P^T ----
      const v8bf* vr = (const v8bf*)&sVt[lrow][kb];
      const v16bf av = join8(vr[h], vr[2 + h]);                  // keys kb+a_koff
      const v16bf bp = join8(((const v8bf*)&sPt[wave][lrow][0])[h], ((const v8bf*)&sPt[wave][lrow][0])[2 + h]);  // keys kb+a_koff
      o = wmma_bf16(av, bp, o);
    }
    // ---- epilogue: l sits at d=8 (VGPR 0 of upper-half lanes) ----
    const float lsum = __shfl(o[0], 16 + lrow, 32);
    const float inv = 1.0f / lsum;
    if (h == 0) {   // d = r (0..7): contiguous per lane -> one 16B store
      v8bf ov;
      #pragma unroll
      for (int r = 0; r < 8; ++r) ov[r] = (__bf16)(o[r] * inv);
      *(v8bf*)(AO + (size_t)(bt * NSEQ + qc0 + lrow) * DMODEL + head * DHEAD) = ov;
    }
  }
}

// ---------------------------------------------------------------------------
// Kernel 3: out = relu(AO @ W10 + b10) @ W11 + b11  (fused, f32 output)
// ---------------------------------------------------------------------------
__global__ __launch_bounds__(256) void out_proj_kernel(
    const __bf16* __restrict__ AO,
    const float* __restrict__ W10, const float* __restrict__ b10,
    const float* __restrict__ W11, const float* __restrict__ b11,
    float* __restrict__ out) {
  __shared__ v16bf sWa[2][4][32];      // 8KB  [kblock][ntile][lane]
  __shared__ v16bf sWb[2][4][32];      // 8KB
  __shared__ __bf16 sZ[8][16 * 64];    // 16KB per-wave intermediate
  const int tid = threadIdx.x;
  {
    __bf16* pa = (__bf16*)sWa;
    __bf16* pb = (__bf16*)sWb;
    for (int idx = tid; idx < 2 * 4 * 32; idx += 256) {
      const int ln = idx & 31, nt = (idx >> 5) & 3, kb = idx >> 7;
      const int hd = ln >> 4, lr = ln & 15;
      const int col = nt * 16 + lr;
      #pragma unroll
      for (int e = 0; e < 16; ++e) {
        const int krow = kb * 32 + (e < 8 ? 8 * hd + e : 8 + 8 * hd + e);
        pa[idx * 16 + e] = (__bf16)W10[krow * DMODEL + col];
        pb[idx * 16 + e] = (__bf16)W11[krow * DMODEL + col];
      }
    }
  }
  __syncthreads();

  const int lane = tid & 31, wave = tid >> 5;
  const int h = lane >> 4, lrow = lane & 15;
  const int row0 = (blockIdx.x * 8 + wave) * 16;

  // GEMM1: Z = relu(AO @ W10 + b10)
  v8f acc[4] = {{}, {}, {}, {}};
  #pragma unroll
  for (int kb = 0; kb < 2; ++kb) {
    const int k0 = kb * 32;
    const v8bf* ar = (const v8bf*)(AO + (size_t)(row0 + lrow) * DMODEL + k0);
    const v16bf a = join8(ar[h], ar[2 + h]);
    #pragma unroll
    for (int nt = 0; nt < 4; ++nt)
      acc[nt] = wmma_bf16(a, sWa[kb][nt][lane], acc[nt]);
  }
  #pragma unroll
  for (int nt = 0; nt < 4; ++nt) {
    const int col = nt * 16 + lrow;
    const float bb = b10[col];
    #pragma unroll
    for (int r = 0; r < 8; ++r)
      sZ[wave][(r + 8 * h) * DMODEL + col] = (__bf16)fmaxf(acc[nt][r] + bb, 0.0f);
  }

  // GEMM2: out = Z @ W11 + b11
  v8f acc2[4] = {{}, {}, {}, {}};
  #pragma unroll
  for (int kb = 0; kb < 2; ++kb) {
    const int k0 = kb * 32;
    const v8bf* zr = (const v8bf*)&sZ[wave][lrow * DMODEL + k0];
    const v16bf a = join8(zr[h], zr[2 + h]);
    #pragma unroll
    for (int nt = 0; nt < 4; ++nt)
      acc2[nt] = wmma_bf16(a, sWb[kb][nt][lane], acc2[nt]);
  }
  #pragma unroll
  for (int nt = 0; nt < 4; ++nt) {
    const int col = nt * 16 + lrow;
    const float bb = b11[col];
    #pragma unroll
    for (int r = 0; r < 8; ++r)
      out[(size_t)(row0 + r + 8 * h) * DMODEL + col] = acc2[nt][r] + bb;
  }
}

// ---------------------------------------------------------------------------
extern "C" void kernel_launch(void* const* d_in, const int* in_sizes, int n_in,
                              void* d_out, int out_size, void* d_ws, size_t ws_size,
                              hipStream_t stream) {
  const float* X   = (const float*)d_in[0];
  const float* STE = (const float*)d_in[1];
  const float* W7  = (const float*)d_in[2];
  const float* b7  = (const float*)d_in[3];
  const float* W8  = (const float*)d_in[4];
  const float* b8  = (const float*)d_in[5];
  const float* W9  = (const float*)d_in[6];
  const float* b9  = (const float*)d_in[7];
  const float* W10 = (const float*)d_in[8];
  const float* b10 = (const float*)d_in[9];
  const float* W11 = (const float*)d_in[10];
  const float* b11 = (const float*)d_in[11];
  float* out = (float*)d_out;

  // workspace: Q,K,V head-major bf16 + AO row-major bf16 (4 x 6 MB = 24 MB)
  const size_t QKV_BYTES = (size_t)NHEAD * NBT * NSEQ * DHEAD * sizeof(__bf16);
  char* ws = (char*)d_ws;
  __bf16* Q  = (__bf16*)(ws);
  __bf16* Kq = (__bf16*)(ws + 1 * QKV_BYTES);
  __bf16* V  = (__bf16*)(ws + 2 * QKV_BYTES);
  __bf16* AO = (__bf16*)(ws + 3 * QKV_BYTES);

  const int gemm_blocks = (NROWS / 16) / 8;   // 384
  qkv_proj_kernel<<<gemm_blocks, 256, 0, stream>>>(X, STE, W7, b7, Q);
  qkv_proj_kernel<<<gemm_blocks, 256, 0, stream>>>(X, STE, W8, b8, Kq);
  qkv_proj_kernel<<<gemm_blocks, 256, 0, stream>>>(X, STE, W9, b9, V);
  attn_kernel<<<NHEAD * NBT, 256, 0, stream>>>(Q, Kq, V, AO);
  out_proj_kernel<<<gemm_blocks, 256, 0, stream>>>(AO, W10, b10, W11, b11, out);
}
